// Model_39676907881874
// MI455X (gfx1250) — matrix operand order repaired and checked by reading, not compiled, not run on hardware
//
#include <hip/hip_runtime.h>

// ---------------------------------------------------------------------------
// Flash-attention forward, B=4,H=16,Sq=Sk=2048,D=DV=128, fp32 in/out,
// reference multiplies QK by sqrt(D). bf16 WMMA (v_wmma_f32_16x16x32_bf16),
// fp32 accumulation, single pass over K/V (online softmax, base-2 exp).
// ---------------------------------------------------------------------------

typedef __attribute__((ext_vector_type(16))) __bf16 v16bf;
typedef __attribute__((ext_vector_type(8)))  __bf16 v8bf;
typedef __attribute__((ext_vector_type(4)))  __bf16 v4bf;
typedef __attribute__((ext_vector_type(8)))  float  v8f;

#define B_   4
#define H_   16
#define SQ_  2048
#define SK_  2048
#define D_   128
#define DV_  128
#define BM   64     // query rows per block
#define BN   64     // key rows per tile step
#define NWAVE 4
#define NTHREADS 128

__device__ __forceinline__ v8bf cvt8(float4 a, float4 b) {
  v8bf r;
  r[0] = (__bf16)a.x; r[1] = (__bf16)a.y; r[2] = (__bf16)a.z; r[3] = (__bf16)a.w;
  r[4] = (__bf16)b.x; r[5] = (__bf16)b.y; r[6] = (__bf16)b.z; r[7] = (__bf16)b.w;
  return r;
}

__device__ __forceinline__ v8f wmma_bf16(v16bf a, v16bf b, v8f c) {
  return __builtin_amdgcn_wmma_f32_16x16x32_bf16(false, a, false, b,
                                                 (short)0, c, false, false);
}

__global__ __launch_bounds__(NTHREADS) void fattn_fwd(
    const float* __restrict__ Q, const float* __restrict__ K,
    const float* __restrict__ V, float* __restrict__ O)
{
  const int bh     = blockIdx.y;        // b*H + h
  const int qblk   = blockIdx.x;
  const int tid    = threadIdx.x;
  const int wave   = tid >> 5;          // owns query rows [wave*16, wave*16+16)
  const int lane   = tid & 31;
  const int lane16 = lane & 15;
  const int half16 = lane >> 4;         // lane-half selects 16-k block (A/B) / M octet (C)

  // reference: softmax(QK * sqrt(128)); fold log2(e) for base-2 exp
  const float c2 = 11.313708498984761f * 1.4426950408889634f;

  const float* Qb = Q + (size_t)bh * SQ_ * D_  + (size_t)qblk * BM * D_;
  const float* Kb = K + (size_t)bh * SK_ * D_;
  const float* Vb = V + (size_t)bh * SK_ * DV_;
  float*       Ob = O + (size_t)bh * SQ_ * DV_ + (size_t)qblk * BM * DV_;

  __shared__ __align__(32) __bf16 lds_k[BN][D_];        // K tile  [key][d]
  __shared__ __align__(32) __bf16 lds_vt[DV_][BN];      // V tile  [dv][key] (transposed)
  __shared__ __align__(32) __bf16 lds_p[NWAVE][16][BN]; // per-wave P relayout

  // ---- persistent Q A-fragments: 4 x (16x32 bf16) ----
  // A element s: k = 32*kk + 16*half + s
  //  -> 16 contiguous elements per fragment at 32*kk + 16*half
  v16bf qa[4];
  {
    const float* qrow = Qb + (size_t)(wave * 16 + lane16) * D_;
#pragma unroll
    for (int kk = 0; kk < 4; ++kk) {
      const float* c0 = qrow + kk * 32 + half16 * 16;
      union { v16bf v; v8bf h[2]; } u;
      u.h[0] = cvt8(*(const float4*)(c0),      *(const float4*)(c0 + 4));
      u.h[1] = cvt8(*(const float4*)(c0 + 8),  *(const float4*)(c0 + 12));
      qa[kk] = u.v;
    }
  }

  const v8f vzero = {0.f, 0.f, 0.f, 0.f, 0.f, 0.f, 0.f, 0.f};
  v8f acc[8];
#pragma unroll
  for (int i = 0; i < 8; ++i) acc[i] = vzero;

  float m_i[8], l_i[8];                 // running max(log2 dom.) / sum, row r+8*half
#pragma unroll
  for (int r = 0; r < 8; ++r) { m_i[r] = -1e30f; l_i[r] = 0.f; }

  for (int kt = 0; kt < SK_ / BN; ++kt) {
    const float* Kt = Kb + (size_t)kt * BN * D_;
    const float* Vt = Vb + (size_t)kt * BN * DV_;

    __syncthreads();   // previous iteration done reading lds_k / lds_vt

    // ---- cooperative stage: K tile fp32 -> bf16 LDS [key][d] ----
    for (int i = tid; i < BN * D_ / 4; i += NTHREADS) {
      int e = i * 4;
      int key = e / D_, d = e % D_;
      float4 f = *(const float4*)(Kt + (size_t)key * D_ + d);
      v4bf p;
      p.x = (__bf16)f.x; p.y = (__bf16)f.y; p.z = (__bf16)f.z; p.w = (__bf16)f.w;
      *(v4bf*)&lds_k[key][d] = p;
    }
    // ---- V tile fp32 -> bf16 LDS, transposed [dv][key] ----
    for (int i = tid; i < BN * DV_ / 4; i += NTHREADS) {
      int e = i * 4;
      int key = e / DV_, dv = e % DV_;
      float4 f = *(const float4*)(Vt + (size_t)key * DV_ + dv);
      lds_vt[dv + 0][key] = (__bf16)f.x;
      lds_vt[dv + 1][key] = (__bf16)f.y;
      lds_vt[dv + 2][key] = (__bf16)f.z;
      lds_vt[dv + 3][key] = (__bf16)f.w;
    }

    // ---- prefetch next K/V tile into cache (global_prefetch_b8) ----
    if (kt + 1 < SK_ / BN) {
      __builtin_prefetch(Kt + BN * D_  + tid * 64, 0, 3);   // 256B/lane -> 32KB
      __builtin_prefetch(Vt + BN * DV_ + tid * 64, 0, 3);
    }

    __syncthreads();

    // ---- S = Q * K^T : 4 WMMA tiles per wave; preload all 4 B-frags/tile ----
    // B element (v,j): d = 32*kk + 16*half + 2*v + j -> 16 contiguous halfs
    v8f s[4];
#pragma unroll
    for (int nt = 0; nt < 4; ++nt) {
      const __bf16* krow = &lds_k[nt * 16 + lane16][0] + half16 * 16;
      v16bf bk0 = *(const v16bf*)(krow);
      v16bf bk1 = *(const v16bf*)(krow + 32);
      v16bf bk2 = *(const v16bf*)(krow + 64);
      v16bf bk3 = *(const v16bf*)(krow + 96);
      v8f c = vzero;
      c = wmma_bf16(qa[0], bk0, c);
      c = wmma_bf16(qa[1], bk1, c);
      c = wmma_bf16(qa[2], bk2, c);
      c = wmma_bf16(qa[3], bk3, c);
      s[nt] = c;
    }

    // ---- online softmax (base-2) on the 16x64 strip ----
#pragma unroll
    for (int r = 0; r < 8; ++r) {
      float mx = m_i[r];
#pragma unroll
      for (int nt = 0; nt < 4; ++nt) {
        s[nt][r] *= c2;                          // into log2 domain
        mx = fmaxf(mx, s[nt][r]);
      }
#pragma unroll
      for (int off = 8; off >= 1; off >>= 1)     // stays within 16-lane group
        mx = fmaxf(mx, __shfl_xor(mx, off, 32));

      float alpha  = exp2f(m_i[r] - mx);
      float rowsum = 0.f;
#pragma unroll
      for (int nt = 0; nt < 4; ++nt) {
        float p = exp2f(s[nt][r] - mx);
        s[nt][r] = p;
        rowsum  += p;
      }
#pragma unroll
      for (int off = 8; off >= 1; off >>= 1)
        rowsum += __shfl_xor(rowsum, off, 32);

      l_i[r] = l_i[r] * alpha + rowsum;
      m_i[r] = mx;
#pragma unroll
      for (int nt = 0; nt < 8; ++nt) acc[nt][r] *= alpha;
    }

    // ---- relayout P: C-layout regs -> LDS row-major -> A-fragments ----
#pragma unroll
    for (int nt = 0; nt < 4; ++nt)
#pragma unroll
      for (int r = 0; r < 8; ++r)
        lds_p[wave][r + 8 * half16][nt * 16 + lane16] = (__bf16)s[nt][r];
    // same-wave DS ops stay ordered; compiler inserts s_wait_dscnt

    v16bf pa[2];
    {
      const __bf16* prow = &lds_p[wave][lane16][0];
#pragma unroll
      for (int kk = 0; kk < 2; ++kk) {
        union { v16bf v; v8bf h[2]; } u;
        u.h[0] = *(const v8bf*)(prow + kk * 32 + half16 * 16);
        u.h[1] = *(const v8bf*)(prow + kk * 32 + 8 + half16 * 16);
        pa[kk] = u.v;
      }
    }

    // ---- O += P * V : 8 dv-tiles x 2 k-chunks; preload both B-frags ----
#pragma unroll
    for (int nt = 0; nt < 8; ++nt) {
      const __bf16* vrow = &lds_vt[nt * 16 + lane16][0] + half16 * 16;
      v16bf bv0 = *(const v16bf*)(vrow);
      v16bf bv1 = *(const v16bf*)(vrow + 32);
      acc[nt] = wmma_bf16(pa[0], bv0, acc[nt]);
      acc[nt] = wmma_bf16(pa[1], bv1, acc[nt]);
    }
  }

  // ---- epilogue: divide by row sum, store fp32 ----
#pragma unroll
  for (int r = 0; r < 8; ++r) {
    float inv = 1.0f / l_i[r];
    int row = wave * 16 + r + 8 * half16;
#pragma unroll
    for (int nt = 0; nt < 8; ++nt)
      Ob[(size_t)row * DV_ + nt * 16 + lane16] = acc[nt][r] * inv;
  }
}

extern "C" void kernel_launch(void* const* d_in, const int* in_sizes, int n_in,
                              void* d_out, int out_size, void* d_ws, size_t ws_size,
                              hipStream_t stream) {
  (void)in_sizes; (void)n_in; (void)out_size; (void)d_ws; (void)ws_size;
  const float* Q = (const float*)d_in[0];
  const float* K = (const float*)d_in[1];
  const float* V = (const float*)d_in[2];
  float*       O = (float*)d_out;
  dim3 grid(SQ_ / BM, B_ * H_);
  dim3 block(NTHREADS);
  hipLaunchKernelGGL(fattn_fwd, grid, block, 0, stream, Q, K, V, O);
}
